// MEGNetBlock_6579889897999
// MI455X (gfx1250) — matrix operand order repaired and checked by reading, not compiled, not run on hardware
//
#include <hip/hip_runtime.h>
#include <hip/hip_bf16.h>

typedef _Float16 hfp;
typedef __attribute__((ext_vector_type(16))) _Float16 v16h;
typedef __attribute__((ext_vector_type(8)))  _Float16 h8;
typedef __attribute__((ext_vector_type(8)))  float    v8f;
typedef __attribute__((ext_vector_type(4)))  float    f4;

// Branch-free stable softplus: max(x,0) + ln2*log2(1 + 2^(-|x|*log2e)).
// log2 argument is in [1,2] -> raw v_log_f32 is safe (no denorm fixup path).
__device__ __forceinline__ float softplus_f(float x) {
    float ax = fabsf(x);
#if __has_builtin(__builtin_amdgcn_exp2f) && __has_builtin(__builtin_amdgcn_logf)
    float t = __builtin_amdgcn_exp2f(ax * -1.442695040888963f);
    return fmaxf(x, 0.f) + 0.6931471805599453f * __builtin_amdgcn_logf(1.f + t);
#else
    return fmaxf(x, 0.f) + __logf(1.f + __expf(-ax));
#endif
}

// ---- WMMA fragment helpers (layouts per CDNA5 ISA 7.12.2, wave32) ----
// A (16xK f16, row-major in LDS): lane L: row = L&15; half h=L>>4 holds K=[8h..8h+7],[16+8h..16+8h+7]
__device__ __forceinline__ v16h ldsA_frag(const hfp* A, int astride, int kbase, int lane) {
    const hfp* p = A + (lane & 15) * astride + kbase + 8 * ((lane >> 4) & 1);
    v16h a;
    *(h8*)&a         = *(const h8*)p;        // 16B aligned
    *(((h8*)&a) + 1) = *(const h8*)(p + 16); // 16B aligned
    return a;
}

// Pack a KxNN row-major f32 weight matrix into fragment-major f16 LDS layout:
// element i of lane l of tile (kt,nt) lives at ((kt*NTL+nt)*32 + l)*16 + i
// so a B-fragment load is one contiguous, 32B-aligned v16h read.
template <int KT, int NTL, int NN>
__device__ __forceinline__ void pack_B(const float* W, hfp* Wp, int tid, int nthreads) {
    const int total = KT * NTL * 512;
    for (int p = tid; p < total; p += nthreads) {
        const int i  = p & 15;
        const int l  = (p >> 4) & 31;
        const int t  = p >> 9;
        const int nt = t % NTL;
        const int kt = t / NTL;
        const int k  = kt * 32 + ((i >> 3) << 4) + ((l >> 4) << 3) + (i & 7); // B: half h holds K=[8h..8h+7],[16+8h..16+8h+7]
        const int n  = nt * 16 + (l & 15);            // col = lane&15
        Wp[p] = (hfp)W[k * NN + n];
    }
}

// Accumulate a 16 x (NT*16) output slab over KT 32-wide K tiles; B is fragment-packed.
// Bias comes from a per-lane register array (bv[nt] = bias[nt*16 + (lane&15)]).
template <int KT, int NT>
__device__ __forceinline__ void gemm_acc(const hfp* Aw, int astride, int abase,
                                         const hfp* Wp, const float* bv,
                                         int lane, v8f* cc) {
#pragma unroll
    for (int nt = 0; nt < NT; ++nt)
#pragma unroll
        for (int i = 0; i < 8; ++i) cc[nt][i] = bv[nt];
#pragma unroll
    for (int kt = 0; kt < KT; ++kt) {
        v16h a = ldsA_frag(Aw, astride, abase + kt * 32, lane);
#pragma unroll
        for (int nt = 0; nt < NT; ++nt) {
            v16h b = *(const v16h*)(Wp + (size_t)((kt * NT + nt) * 32 + lane) * 16);
            cc[nt] = __builtin_amdgcn_wmma_f32_16x16x32_f16(false, a, false, b,
                                                            (short)0, cc[nt], false, false);
        }
    }
}

// Store softplus(acc) to the LDS activation slab. D layout: row = 8h+i, col = lane&15.
template <int NT>
__device__ __forceinline__ void store_act(hfp* Aw, int astride, int obase, int lane, const v8f* cc) {
    const int col = lane & 15;
    const int h   = (lane >> 4) & 1;
#pragma unroll
    for (int nt = 0; nt < NT; ++nt)
#pragma unroll
        for (int i = 0; i < 8; ++i)
            Aw[(8 * h + i) * astride + obase + nt * 16 + col] = (hfp)softplus_f(cc[nt][i]);
}

// Convert 32 consecutive f32 -> 32 consecutive f16 in LDS with wide ops.
__device__ __forceinline__ void cvt_store32(hfp* dst, const float* src) {
    const f4* s4 = (const f4*)src;
#pragma unroll
    for (int q = 0; q < 4; ++q) {
        f4 a = s4[2 * q], b = s4[2 * q + 1];
        h8 v;
        v[0] = (hfp)a[0]; v[1] = (hfp)a[1]; v[2] = (hfp)a[2]; v[3] = (hfp)a[3];
        v[4] = (hfp)b[0]; v[5] = (hfp)b[1]; v[6] = (hfp)b[2]; v[7] = (hfp)b[3];
        *(h8*)(dst + 8 * q) = v;
    }
}
__device__ __forceinline__ void cvt_store32_scaled(hfp* dst, const float* src, float s) {
    const f4* s4 = (const f4*)src;
#pragma unroll
    for (int q = 0; q < 4; ++q) {
        f4 a = s4[2 * q], b = s4[2 * q + 1];
        h8 v;
        v[0] = (hfp)(a[0] * s); v[1] = (hfp)(a[1] * s); v[2] = (hfp)(a[2] * s); v[3] = (hfp)(a[3] * s);
        v[4] = (hfp)(b[0] * s); v[5] = (hfp)(b[1] * s); v[6] = (hfp)(b[2] * s); v[7] = (hfp)(b[3] * s);
        *(h8*)(dst + 8 * q) = v;
    }
}

// ---------------------------------------------------------------------
__global__ void k_zero(float* p, size_t n) {
    size_t i = (size_t)blockIdx.x * 256u + threadIdx.x;
    if (i < n) p[i] = 0.f;
}

// s = softplus(state_feat @ bs_W + bs_b)   [32,64] -- tiny, scalar
__global__ void k_state_block(const float* state_feat, const float* bsW, const float* bsb,
                              float* s_buf) {
    for (int idx = threadIdx.x; idx < 32 * 64; idx += blockDim.x) {
        int g = idx >> 6, d = idx & 63;
        float acc = bsb[d];
        for (int k = 0; k < 64; ++k) acc += state_feat[g * 64 + k] * bsW[k * 64 + d];
        s_buf[idx] = softplus_f(acc);
    }
}

// n = softplus(node_feat @ bn_W + bn_b)   [N,64] -- WMMA, 8 waves x 16 rows / block
__global__ void k_node_block(const float* node_feat, const float* bnW, const float* bnb,
                             float* n_buf, int N) {
    __shared__ hfp   sW[64 * 64];
    __shared__ float sB[64];
    __shared__ hfp   sA[8 * 16 * 72];
    const int tid = threadIdx.x, w = tid >> 5, lane = tid & 31;
    pack_B<2, 4, 64>(bnW, sW, tid, 256);
    if (tid < 64) sB[tid] = bnb[tid];
    __syncthreads();

    const int col = lane & 15, h = (lane >> 4) & 1;
    float rB[4];
#pragma unroll
    for (int nt = 0; nt < 4; ++nt) rB[nt] = sB[nt * 16 + col];

    hfp* Aw = sA + w * 16 * 72;
    const int r0 = blockIdx.x * 128 + w * 16;
    {
        int j = lane >> 1, hf = lane & 1;
        int r = r0 + j; if (r >= N) r = N - 1;
        cvt_store32(Aw + j * 72 + hf * 32, node_feat + (size_t)r * 64 + hf * 32);
    }
    __syncthreads();
    v8f cc[4];
    gemm_acc<2, 4>(Aw, 72, 0, sW, rB, lane, cc);
#pragma unroll
    for (int nt = 0; nt < 4; ++nt)
#pragma unroll
        for (int i = 0; i < 8; ++i) {
            int r = r0 + 8 * h + i;
            if (r < N) n_buf[(size_t)r * 64 + nt * 16 + col] = softplus_f(cc[nt][i]);
        }
}

// ---- fused edge pipeline: e-block MLP + concat + 256->128->128->64 MLP + scatters ----
#define AST 264
__global__ void k_edge(const float* __restrict__ edge_feat, const int* __restrict__ ei,
                       const int* __restrict__ batch, const float* __restrict__ s_buf,
                       const float* __restrict__ n_buf,
                       const float* beW, const float* beb,
                       const float* W0, const float* b0, const float* W1, const float* b1,
                       const float* W2, const float* b2,
                       float* out_edge, float* ve_sum, float* ve_cnt,
                       float* ue_sum, float* ue_cnt, int E) {
    extern __shared__ char smem[];
    hfp*   sBeW  = (hfp*)smem;                 //  4096 h (frag-packed)
    hfp*   sW0   = (hfp*)(smem + 8192);        // 32768 h
    hfp*   sW1   = (hfp*)(smem + 73728);       // 16384 h
    hfp*   sW2   = (hfp*)(smem + 106496);      //  8192 h
    float* sBb   = (float*)(smem + 122880);    //    64
    float* sB0   = sBb + 64;                   //   128
    float* sB1   = sB0 + 128;                  //   128
    float* sB2   = sB1 + 128;                  //    64
    float* sGsum = (float*)(smem + 124416);    //  2048
    float* sGcnt = (float*)(smem + 132608);    //    32
    int*   sIdx  = (int*)(smem + 132736);      // 8*48
    hfp*   sA    = (hfp*)(smem + 134272);      // 8*16*264 h

    const int tid = threadIdx.x, w = tid >> 5, lane = tid & 31;
    pack_B<2, 4, 64>(beW, sBeW, tid, 256);
    pack_B<8, 8, 128>(W0, sW0, tid, 256);
    pack_B<4, 8, 128>(W1, sW1, tid, 256);
    pack_B<4, 4, 64>(W2, sW2, tid, 256);
    if (tid < 64)  sBb[tid] = beb[tid];
    if (tid < 128) { sB0[tid] = b0[tid]; sB1[tid] = b1[tid]; }
    if (tid < 64)  sB2[tid] = b2[tid];
    for (int i = tid; i < 2048; i += 256) sGsum[i] = 0.f;
    if (tid < 32) sGcnt[tid] = 0.f;
    __syncthreads();

    const int col = lane & 15, h = (lane >> 4) & 1;
    // hoist per-lane bias values into registers (invariant across chunks)
    float rBe[4], rB0[8], rB1[8], rB2[4];
#pragma unroll
    for (int nt = 0; nt < 4; ++nt) rBe[nt] = sBb[nt * 16 + col];
#pragma unroll
    for (int nt = 0; nt < 8; ++nt) rB0[nt] = sB0[nt * 16 + col];
#pragma unroll
    for (int nt = 0; nt < 8; ++nt) rB1[nt] = sB1[nt * 16 + col];
#pragma unroll
    for (int nt = 0; nt < 4; ++nt) rB2[nt] = sB2[nt * 16 + col];

    hfp* Aw  = sA + w * (16 * AST);
    int* idxw = sIdx + w * 48;
    const int nchunk = (E + 127) >> 7;

    for (int chunk = blockIdx.x; chunk < nchunk; chunk += gridDim.x) {
        const int e0 = chunk * 128 + w * 16;
        if (lane < 16) {
            int e = e0 + lane; if (e >= E) e = E - 1;
            int s_ = ei[e];
            idxw[lane]      = ei[E + e];   // dst
            idxw[16 + lane] = batch[s_];   // graph of src
            idxw[32 + lane] = s_;          // src
        }
        __syncthreads();
        {   // stage the [16 x 256] concat A-tile (f16): [n[src] | n[dst] | efeat | s[g]]
            int j = lane >> 1, hf = lane & 1;
            int ee = e0 + j; if (ee >= E) ee = E - 1;
            int sj = idxw[32 + j], dj = idxw[j], gj = idxw[16 + j];
            hfp* ar = Aw + j * AST + hf * 32;
            cvt_store32(ar,       n_buf + (size_t)sj * 64 + hf * 32);
            cvt_store32(ar + 64,  n_buf + (size_t)dj * 64 + hf * 32);
            cvt_store32(ar + 128, edge_feat + (size_t)ee * 64 + hf * 32);
            cvt_store32(ar + 192, s_buf + gj * 64 + hf * 32);
        }
        __syncthreads();
        {   // edge block MLP: e = softplus(efeat @ be_W + be_b), in-place into cols 128..191
            v8f cc[4];
            gemm_acc<2, 4>(Aw, AST, 128, sBeW, rBe, lane, cc);
            __syncthreads();
            store_act<4>(Aw, AST, 128, lane, cc);
            __syncthreads();
        }
        {   // layer0: 256 -> 128, act1 overwrites cols 0..127
            v8f cc[8];
            gemm_acc<8, 8>(Aw, AST, 0, sW0, rB0, lane, cc);
            __syncthreads();
            store_act<8>(Aw, AST, 0, lane, cc);
            __syncthreads();
        }
        {   // layer1: 128 -> 128, act2 overwrites cols 0..127
            v8f cc[8];
            gemm_acc<4, 8>(Aw, AST, 0, sW1, rB1, lane, cc);
            __syncthreads();
            store_act<8>(Aw, AST, 0, lane, cc);
            __syncthreads();
        }
        {   // layer2: 128 -> 64, write skip output + scatter
            v8f cc[4];
            gemm_acc<4, 4>(Aw, AST, 0, sW2, rB2, lane, cc);
#pragma unroll
            for (int nt = 0; nt < 4; ++nt)
#pragma unroll
                for (int i = 0; i < 8; ++i) {
                    const int m = 8 * h + i;
                    const int e = e0 + m;
                    if (e < E) {
                        float v = softplus_f(cc[nt][i]);
                        const int c = nt * 16 + col;
                        size_t o = (size_t)e * 64 + c;
                        out_edge[o] = v + edge_feat[o];
                        atomicAdd(&ve_sum[(size_t)idxw[m] * 64 + c], v);
                        atomicAdd(&sGsum[idxw[16 + m] * 64 + c], v);
                    }
                }
            if (lane < 16 && (e0 + lane) < E) {
                atomicAdd(&ve_cnt[idxw[lane]], 1.f);
                atomicAdd(&sGcnt[idxw[16 + lane]], 1.f);
            }
            __syncthreads();
        }
    }
    __syncthreads();
    for (int i = tid; i < 2048; i += 256) atomicAdd(&ue_sum[i], sGsum[i]);
    if (tid < 32) atomicAdd(&ue_cnt[tid], sGcnt[tid]);
}

// ---- node update MLP: [n | ve_mean | s[batch]] (192) -> 128 -> 128 -> 64 ----
#define NST 200
__global__ void k_node_upd(const float* __restrict__ node_feat, const int* __restrict__ batch,
                           const float* __restrict__ s_buf, const float* __restrict__ n_buf,
                           const float* __restrict__ ve_sum, const float* __restrict__ ve_cnt,
                           const float* W0, const float* b0, const float* W1, const float* b1,
                           const float* W2, const float* b2,
                           float* out_node, float* uv_sum, float* uv_cnt, int N) {
    extern __shared__ char smem[];
    hfp*   sW0n  = (hfp*)smem;                 // 24576 h (frag-packed)
    hfp*   sW1n  = (hfp*)(smem + 49152);       // 16384 h
    hfp*   sW2n  = (hfp*)(smem + 81920);       //  8192 h
    float* sB0   = (float*)(smem + 98304);
    float* sB1   = sB0 + 128;
    float* sB2   = sB1 + 128;
    float* sGsum = (float*)(smem + 99584);
    float* sGcnt = (float*)(smem + 107776);
    int*   sIdx  = (int*)(smem + 107904);
    hfp*   sA    = (hfp*)(smem + 108416);      // 8*16*200 h

    const int tid = threadIdx.x, w = tid >> 5, lane = tid & 31;
    pack_B<6, 8, 128>(W0, sW0n, tid, 256);
    pack_B<4, 8, 128>(W1, sW1n, tid, 256);
    pack_B<4, 4, 64>(W2, sW2n, tid, 256);
    if (tid < 128) { sB0[tid] = b0[tid]; sB1[tid] = b1[tid]; }
    if (tid < 64)  sB2[tid] = b2[tid];
    for (int i = tid; i < 2048; i += 256) sGsum[i] = 0.f;
    if (tid < 32) sGcnt[tid] = 0.f;
    __syncthreads();

    const int col = lane & 15, h = (lane >> 4) & 1;
    float rB0[8], rB1[8], rB2[4];
#pragma unroll
    for (int nt = 0; nt < 8; ++nt) rB0[nt] = sB0[nt * 16 + col];
#pragma unroll
    for (int nt = 0; nt < 8; ++nt) rB1[nt] = sB1[nt * 16 + col];
#pragma unroll
    for (int nt = 0; nt < 4; ++nt) rB2[nt] = sB2[nt * 16 + col];

    hfp* Aw  = sA + w * (16 * NST);
    int* idxw = sIdx + w * 16;
    const int r0 = blockIdx.x * 128 + w * 16;
    if (lane < 16) {
        int r = r0 + lane; if (r >= N) r = N - 1;
        idxw[lane] = batch[r];
    }
    __syncthreads();
    {
        int j = lane >> 1, hf = lane & 1;
        int r = r0 + j; if (r >= N) r = N - 1;
        int g = idxw[j];
        float inv = 1.f / fmaxf(ve_cnt[r], 1.f);
        hfp* ar = Aw + j * NST + hf * 32;
        cvt_store32(ar,        n_buf + (size_t)r * 64 + hf * 32);
        cvt_store32_scaled(ar + 64, ve_sum + (size_t)r * 64 + hf * 32, inv);
        cvt_store32(ar + 128,  s_buf + g * 64 + hf * 32);
    }
    __syncthreads();
    {
        v8f cc[8];
        gemm_acc<6, 8>(Aw, NST, 0, sW0n, rB0, lane, cc);
        __syncthreads();
        store_act<8>(Aw, NST, 0, lane, cc);
        __syncthreads();
    }
    {
        v8f cc[8];
        gemm_acc<4, 8>(Aw, NST, 0, sW1n, rB1, lane, cc);
        __syncthreads();
        store_act<8>(Aw, NST, 0, lane, cc);
        __syncthreads();
    }
    {
        v8f cc[4];
        gemm_acc<4, 4>(Aw, NST, 0, sW2n, rB2, lane, cc);
#pragma unroll
        for (int nt = 0; nt < 4; ++nt)
#pragma unroll
            for (int i = 0; i < 8; ++i) {
                const int m = 8 * h + i;
                const int r = r0 + m;
                if (r < N) {
                    float v = softplus_f(cc[nt][i]);
                    const int c = nt * 16 + col;
                    size_t o = (size_t)r * 64 + c;
                    out_node[o] = v + node_feat[o];
                    atomicAdd(&sGsum[idxw[m] * 64 + c], v);
                }
            }
        if (lane < 16 && (r0 + lane) < N) atomicAdd(&sGcnt[idxw[lane]], 1.f);
    }
    __syncthreads();
    for (int i = tid; i < 2048; i += 256) atomicAdd(&uv_sum[i], sGsum[i]);
    if (tid < 32) atomicAdd(&uv_cnt[tid], sGcnt[tid]);
}

// ---- state update MLP (tiny, scalar f32): [s | ue_mean | uv_mean] -> 128 -> 128 -> 64 + skip ----
__global__ void k_state_upd(const float* state_feat, const float* s_buf,
                            const float* ue_sum, const float* ue_cnt,
                            const float* uv_sum, const float* uv_cnt,
                            const float* W0, const float* b0, const float* W1, const float* b1,
                            const float* W2, const float* b2, float* out_state) {
    __shared__ float si[32 * 192];
    __shared__ float h1[32 * 128];
    __shared__ float h2[32 * 128];
    const int tid = threadIdx.x;
    for (int idx = tid; idx < 2048; idx += 256) {
        int g = idx >> 6, d = idx & 63;
        si[g * 192 + d]       = s_buf[idx];
        si[g * 192 + 64 + d]  = ue_sum[idx] / fmaxf(ue_cnt[g], 1.f);
        si[g * 192 + 128 + d] = uv_sum[idx] / fmaxf(uv_cnt[g], 1.f);
    }
    __syncthreads();
    for (int idx = tid; idx < 32 * 128; idx += 256) {
        int g = idx >> 7, j = idx & 127;
        float a = b0[j];
        for (int k = 0; k < 192; ++k) a += si[g * 192 + k] * W0[k * 128 + j];
        h1[idx] = softplus_f(a);
    }
    __syncthreads();
    for (int idx = tid; idx < 32 * 128; idx += 256) {
        int g = idx >> 7, j = idx & 127;
        float a = b1[j];
        for (int k = 0; k < 128; ++k) a += h1[g * 128 + k] * W1[k * 128 + j];
        h2[idx] = softplus_f(a);
    }
    __syncthreads();
    for (int idx = tid; idx < 2048; idx += 256) {
        int g = idx >> 6, j = idx & 63;
        float a = b2[j];
        for (int k = 0; k < 128; ++k) a += h2[g * 128 + k] * W2[k * 64 + j];
        out_state[idx] = softplus_f(a) + state_feat[idx];
    }
}

extern "C" void kernel_launch(void* const* d_in, const int* in_sizes, int n_in,
                              void* d_out, int out_size, void* d_ws, size_t ws_size,
                              hipStream_t stream) {
    const float* edge_feat  = (const float*)d_in[0];
    const float* node_feat  = (const float*)d_in[1];
    const float* state_feat = (const float*)d_in[2];
    const int*   edge_index = (const int*)d_in[3];
    const int*   batch      = (const int*)d_in[4];
    const float* beW  = (const float*)d_in[5];  const float* beb  = (const float*)d_in[6];
    const float* bnW  = (const float*)d_in[7];  const float* bnb  = (const float*)d_in[8];
    const float* bsW  = (const float*)d_in[9];  const float* bsb  = (const float*)d_in[10];
    const float* ceW0 = (const float*)d_in[11]; const float* ceb0 = (const float*)d_in[12];
    const float* ceW1 = (const float*)d_in[13]; const float* ceb1 = (const float*)d_in[14];
    const float* ceW2 = (const float*)d_in[15]; const float* ceb2 = (const float*)d_in[16];
    const float* cnW0 = (const float*)d_in[17]; const float* cnb0 = (const float*)d_in[18];
    const float* cnW1 = (const float*)d_in[19]; const float* cnb1 = (const float*)d_in[20];
    const float* cnW2 = (const float*)d_in[21]; const float* cnb2 = (const float*)d_in[22];
    const float* csW0 = (const float*)d_in[23]; const float* csb0 = (const float*)d_in[24];
    const float* csW1 = (const float*)d_in[25]; const float* csb1 = (const float*)d_in[26];
    const float* csW2 = (const float*)d_in[27]; const float* csb2 = (const float*)d_in[28];

    const int E = in_sizes[0] / 64;  // 800000
    const int N = in_sizes[1] / 64;  // 50000

    // workspace layout (floats): [ve_sum | ve_cnt | ue_sum | ue_cnt | uv_sum | uv_cnt | s_buf | n_buf]
    float* ws     = (float*)d_ws;
    float* ve_sum = ws;
    float* ve_cnt = ve_sum + (size_t)N * 64;
    float* ue_sum = ve_cnt + N;
    float* ue_cnt = ue_sum + 2048;
    float* uv_sum = ue_cnt + 32;
    float* uv_cnt = uv_sum + 2048;
    float* s_buf  = uv_cnt + 32;
    float* n_buf  = s_buf + 2048;

    float* out_edge  = (float*)d_out;
    float* out_node  = out_edge + (size_t)E * 64;
    float* out_state = out_node + (size_t)N * 64;

    const size_t zn = (size_t)N * 64 + N + 2048 + 32 + 2048 + 32;
    k_zero<<<(unsigned)((zn + 255) / 256), 256, 0, stream>>>(ws, zn);
    k_state_block<<<1, 256, 0, stream>>>(state_feat, bsW, bsb, s_buf);
    k_node_block<<<(N + 127) / 128, 256, 0, stream>>>(node_feat, bnW, bnb, n_buf, N);
    k_edge<<<1250, 256, 201856, stream>>>(edge_feat, edge_index, batch, s_buf, n_buf,
                                          beW, beb, ceW0, ceb0, ceW1, ceb1, ceW2, ceb2,
                                          out_edge, ve_sum, ve_cnt, ue_sum, ue_cnt, E);
    k_node_upd<<<(N + 127) / 128, 256, 159616, stream>>>(node_feat, batch, s_buf, n_buf,
                                                         ve_sum, ve_cnt,
                                                         cnW0, cnb0, cnW1, cnb1, cnW2, cnb2,
                                                         out_node, uv_sum, uv_cnt, N);
    k_state_upd<<<1, 256, 0, stream>>>(state_feat, s_buf, ue_sum, ue_cnt, uv_sum, uv_cnt,
                                       csW0, csb0, csW1, csb1, csW2, csb2, out_state);
}
